// RvNN_80444737454318
// MI455X (gfx1250) — matrix operand order repaired and checked by reading, not compiled, not run on hardware
//
#include <hip/hip_runtime.h>

// ---------------------------------------------------------------- types ----
typedef _Float16 half_t;
typedef __attribute__((ext_vector_type(16))) _Float16 v16h;
typedef __attribute__((ext_vector_type(2)))  _Float16 h2;
typedef __attribute__((ext_vector_type(8)))  float    v8f;

#define HID   1024
#define NNODE 4096
#define VOCAB 30000
#define WPN   32
#define NLEAF 1024
#define SCAN_BLOCKS 64            // == HID/16: one 16-column slice per block

// ------------------------------------------------- WMMA fragment loaders ---
// 16-bit A matrix 16x32 (MxK), contiguous k = 16h + s order used here (B matches):
//   lanes 0-15: M=lane,   VGPR v: K = 2v          (+16 for lanes 16-31)
__device__ __forceinline__ v16h load_a_frag(const half_t* row, int k0, int hf) {
  v16h a;
#pragma unroll
  for (int v = 0; v < 8; ++v) {
    int kb = 2 * v + 16 * hf;
    h2 p = *(const h2*)(row + k0 + kb);
    a[2 * v] = p.x; a[2 * v + 1] = p.y;
  }
  return a;
}

// 16-bit B matrix 32x16 (KxN): lane gives N (lanes 16-31: K+16),
// VGPR v holds K pair {2v, 2v+1}.  B[k][n] = W[n][k]  (x @ W^T).
// Works for both global and LDS source pointers (generic addrspace).
__device__ __forceinline__ v16h load_b_frag(const half_t* wrow, int k0, int hf) {
  v16h b;
#pragma unroll
  for (int v = 0; v < 8; ++v) {
    h2 p = *(const h2*)(wrow + k0 + 2 * v + 16 * hf);
    b[2 * v] = p.x; b[2 * v + 1] = p.y;
  }
  return b;
}

#define WMMA_F16(a, b, c) \
  __builtin_amdgcn_wmma_f32_16x16x32_f16(false, (a), false, (b), (short)0, (c), false, false)

// -------------------------------------------------------------- kernels ----

// Zero root-parent hidden row + one-shot barrier slots (every launch: ws is
// poisoned before timing and barriers are left non-zero after each call).
__global__ void init_ws_kernel(float* node_h, half_t* node_h16, unsigned* bars) {
  int i = blockIdx.x * blockDim.x + threadIdx.x;
  if (i < HID) { node_h[i] = 0.f; node_h16[i] = (half_t)0.f; }
  if (i < 16384) bars[i] = 0u;
}

// f32 -> f16 conversion for the six HxH matrices.
__global__ void cvt_weights_kernel(const float* Wz, const float* Wr, const float* Wh,
                                   const float* Uz, const float* Ur, const float* Uh,
                                   half_t* W16, half_t* U16) {
  const int M = HID * HID;
  for (int i = blockIdx.x * blockDim.x + threadIdx.x; i < M;
       i += gridDim.x * blockDim.x) {
    W16[i]         = (half_t)Wz[i];
    W16[M + i]     = (half_t)Wr[i];
    W16[2 * M + i] = (half_t)Wh[i];
    U16[i]         = (half_t)Uz[i];
    U16[M + i]     = (half_t)Ur[i];
    U16[2 * M + i] = (half_t)Uh[i];
  }
}

// Weighted bag-of-words embedding: xe16[n,h] = sum_l w[l]*E[h, idx[l]].
// E (122.9MB) fits the 192MB L2, so the column gather is L2-resident.
__global__ void embed_kernel(const float* __restrict__ xw, const int* __restrict__ xi,
                             const float* __restrict__ E, half_t* __restrict__ xe16) {
  __shared__ float w[WPN];
  __shared__ int   idx[WPN];
  int n = blockIdx.x;
  if (threadIdx.x < WPN) {
    w[threadIdx.x]   = xw[n * WPN + threadIdx.x];
    idx[threadIdx.x] = xi[n * WPN + threadIdx.x];
  }
  __syncthreads();
  for (int h = threadIdx.x; h < HID; h += blockDim.x) {
    float acc = 0.f;
#pragma unroll 4
    for (int l = 0; l < WPN; ++l)
      acc += w[l] * E[(size_t)h * VOCAB + idx[l]];
    xe16[(size_t)n * HID + h] = (half_t)acc;
  }
}

// wzrh[g][m][o] = sum_k xe[m,k] * W_g[o,k] + b_g[o]  (g = z,r,h fused: 3072 cols)
// 32x32 register tile per wave (2 A-frags x 2 B-frags -> 4 wmma per K-step,
// 16 FLOP/B from cache).  The 8 waves of a block share one 32-column group so
// B rows hit in L1 across waves.
__global__ void __launch_bounds__(256) gemm_xew_kernel(
    const half_t* __restrict__ xe16, const half_t* __restrict__ W16,
    const float* __restrict__ bz, const float* __restrict__ br,
    const float* __restrict__ bh, float* __restrict__ wzrh) {
  const int MT = NNODE / 32;                       // 128 row tiles
  int wave = threadIdx.x >> 5, lane = threadIdx.x & 31;
  int nt = blockIdx.x / (MT / 8);                  // 32-col group (0..95)
  int mt = (blockIdx.x % (MT / 8)) * 8 + wave;     // 32-row tile
  int hf = lane >> 4, nl = lane & 15;
  int g = (nt * 32) >> 10;                         // matrix id (groups of 32
  int oc0 = (nt * 32 + nl) & (HID - 1);            //  cols never straddle g)
  int oc1 = oc0 + 16;
  const half_t* arow0 = xe16 + (size_t)(mt * 32 + nl) * HID;
  const half_t* arow1 = arow0 + (size_t)16 * HID;
  const half_t* brow0 = W16 + (size_t)g * HID * HID + (size_t)oc0 * HID;
  const half_t* brow1 = W16 + (size_t)g * HID * HID + (size_t)oc1 * HID;
  v8f a00 = {}, a01 = {}, a10 = {}, a11 = {};
  for (int k0 = 0; k0 < HID; k0 += 32) {
    __builtin_prefetch(arow0 + k0 + 64, 0, 1);     // global_prefetch_b8
    v16h fa0 = load_a_frag(arow0, k0, hf);
    v16h fa1 = load_a_frag(arow1, k0, hf);
    v16h fb0 = load_b_frag(brow0, k0, hf);
    v16h fb1 = load_b_frag(brow1, k0, hf);
    a00 = WMMA_F16(fa0, fb0, a00);
    a01 = WMMA_F16(fa0, fb1, a01);
    a10 = WMMA_F16(fa1, fb0, a10);
    a11 = WMMA_F16(fa1, fb1, a11);
  }
  const float* bias = (g == 0) ? bz : (g == 1) ? br : bh;
  float bv0 = bias[oc0], bv1 = bias[oc1];
  float* ob = wzrh + (size_t)g * NNODE * HID;
#pragma unroll
  for (int j = 0; j < 8; ++j) {
    int m0 = mt * 32 + j + (hf ? 8 : 0);           // C/D row layout
    int m1 = m0 + 16;
    ob[(size_t)m0 * HID + oc0] = a00[j] + bv0;
    ob[(size_t)m0 * HID + oc1] = a01[j] + bv1;
    ob[(size_t)m1 * HID + oc0] = a10[j] + bv0;
    ob[(size_t)m1 * HID + oc1] = a11[j] + bv1;
  }
}

// Topological level build (random recursive tree -> ~25 levels).  Tiny
// sequential pass; parent[i] <= i guarantees order.  Levels padded to 16
// with -1 sentinels so every WMMA m-tile is full.
__global__ void build_levels_kernel(const int* __restrict__ parent, int* depth,
                                    int* level_n, int* level_off, int* cursor,
                                    int* nodes_sorted, int* nlevels_p) {
  if (threadIdx.x | blockIdx.x) return;
  int nlev = 0;
  for (int i = 0; i < NNODE; ++i) {
    int p = parent[i];                             // row index into node_h
    int d = (p == 0) ? 0 : (depth[p - 1] + 1);
    depth[i] = d;
    if (d + 1 > nlev) nlev = d + 1;
  }
  for (int l = 0; l < nlev; ++l) level_n[l] = 0;
  for (int i = 0; i < NNODE; ++i) level_n[depth[i]]++;
  int off = 0;
  for (int l = 0; l < nlev; ++l) {
    level_off[l] = off;
    cursor[l] = off;
    off += ((level_n[l] + 15) >> 4) << 4;
  }
  for (int j = 0; j < off; ++j) nodes_sorted[j] = -1;
  for (int i = 0; i < NNODE; ++i) nodes_sorted[cursor[depth[i]]++] = i;
  *nlevels_p = nlev;
}

// One-shot grid barrier (fresh slot per use; slots zeroed by init kernel).
__device__ __forceinline__ void grid_barrier(unsigned* slot, int nblocks) {
  __syncthreads();
  if (threadIdx.x == 0) {
    __threadfence();
    atomicAdd(slot, 1u);
    volatile unsigned* vs = slot;
    while (*vs < (unsigned)nblocks) __builtin_amdgcn_s_sleep(1);
    __threadfence();
  }
  __syncthreads();
}

// Persistent level-parallel tree-GRU scan with LDS-resident recurrent weights.
// Block b owns output columns [b*16, b*16+16); the Uz/Ur/Uh rows for those
// columns (3 x 16 x 1024 f16 = 96KB, inside the 320KB WGP LDS) are staged
// once and reused across ALL levels and m-tiles -> U read from HBM/L2 once.
// Per level:
//   phase A: [z|r]acc = PH @ Uz^T / Ur^T (WMMA, B from LDS), stash z, r*ph.
//   phase B: acc = RPH @ Uh^T, c = tanh(wh+acc), h' = z*ph + (1-z)*c, commit.
__global__ void __launch_bounds__(256) tree_scan_kernel(
    const int* __restrict__ parent, const float* __restrict__ wzrh,
    const half_t* __restrict__ U16, float* __restrict__ node_h,
    half_t* __restrict__ node_h16, half_t* __restrict__ rph16,
    float* __restrict__ zbuf, const int* __restrict__ level_n,
    const int* __restrict__ level_off, const int* __restrict__ nodes_sorted,
    const int* __restrict__ nlevels_p, unsigned* bars, int nblocks) {
  extern __shared__ half_t Us[];                   // [3][16][HID]
  const float* wz = wzrh;
  const float* wr = wzrh + (size_t)NNODE * HID;
  const float* wh = wzrh + (size_t)2 * NNODE * HID;

  int bcol = blockIdx.x * 16;                      // owned column slice
  for (int i = threadIdx.x; i < 16 * HID; i += blockDim.x) {
    int c = i >> 10, k = i & (HID - 1);
    size_t src = (size_t)(bcol + c) * HID + k;
    Us[i]                = U16[src];                               // Uz
    Us[16 * HID + i]     = U16[(size_t)HID * HID + src];           // Ur
    Us[32 * HID + i]     = U16[(size_t)2 * HID * HID + src];       // Uh
  }
  __syncthreads();

  int wave = threadIdx.x >> 5, lane = threadIdx.x & 31;
  int hf = lane >> 4, nl = lane & 15;
  int oc = bcol + nl;
  const half_t* bz_l = Us + (size_t)nl * HID;
  const half_t* br_l = Us + (size_t)(16 + nl) * HID;
  const half_t* bh_l = Us + (size_t)(32 + nl) * HID;
  int nlev = *nlevels_p;

  for (int lev = 0; lev < nlev; ++lev) {
    int n = level_n[lev], off = level_off[lev];
    int mtiles = (n + 15) >> 4;

    // ---- phase A: gates z, r --------------------------------------------
    for (int mt = wave; mt < mtiles; mt += 8) {
      int nodeA = nodes_sorted[off + mt * 16 + nl];
      int prowA = 0;
      if (nodeA >= 0) prowA = parent[nodeA];       // row 0 == zeros for pads
      const half_t* arow = node_h16 + (size_t)prowA * HID;
      v8f zacc = {}, racc = {};
      for (int k0 = 0; k0 < HID; k0 += 32) {
        v16h a  = load_a_frag(arow, k0, hf);
        v16h b1 = load_b_frag(bz_l, k0, hf);       // ds_load
        zacc = WMMA_F16(a, b1, zacc);
        v16h b2 = load_b_frag(br_l, k0, hf);
        racc = WMMA_F16(a, b2, racc);
      }
#pragma unroll
      for (int j = 0; j < 8; ++j) {
        int pos = mt * 16 + j + (hf ? 8 : 0);
        int node = nodes_sorted[off + pos];
        if (node < 0) continue;
        int prow = parent[node];
        float phv = node_h[(size_t)prow * HID + oc];
        float z = 1.f / (1.f + __expf(-(wz[(size_t)node * HID + oc] + zacc[j])));
        float r = 1.f / (1.f + __expf(-(wr[(size_t)node * HID + oc] + racc[j])));
        rph16[(size_t)pos * HID + oc] = (half_t)(r * phv);
        zbuf[(size_t)pos * HID + oc] = z;
      }
    }
    grid_barrier(&bars[2 * lev], nblocks);

    // ---- phase B: candidate + commit ------------------------------------
    for (int mt = wave; mt < mtiles; mt += 8) {
      const half_t* arow = rph16 + (size_t)(mt * 16 + nl) * HID;
      v8f acc = {};
      for (int k0 = 0; k0 < HID; k0 += 32) {
        v16h a = load_a_frag(arow, k0, hf);
        v16h b = load_b_frag(bh_l, k0, hf);
        acc = WMMA_F16(a, b, acc);
      }
#pragma unroll
      for (int j = 0; j < 8; ++j) {
        int pos = mt * 16 + j + (hf ? 8 : 0);
        int node = nodes_sorted[off + pos];
        if (node < 0) continue;
        int prow = parent[node];
        float phv = node_h[(size_t)prow * HID + oc];
        float c = tanhf(wh[(size_t)node * HID + oc] + acc[j]);
        float z = zbuf[(size_t)pos * HID + oc];
        float hn = z * phv + (1.f - z) * c;
        node_h[(size_t)(node + 1) * HID + oc] = hn;
        node_h16[(size_t)(node + 1) * HID + oc] = (half_t)hn;
      }
    }
    grid_barrier(&bars[2 * lev + 1], nblocks);
  }
}

// Leaf max-pool -> 4-way softmax -> squared-error loss.  out = [pred(4), loss]
__global__ void finalize_kernel(const float* __restrict__ node_h,
                                const int* __restrict__ leaf_idxs,
                                const float* __restrict__ Wout,
                                const float* __restrict__ bout,
                                const float* __restrict__ y, float* out) {
  __shared__ float fs[HID];
  __shared__ float red[HID];
  __shared__ int   li[NLEAF];
  __shared__ float logits[4];
  int h = threadIdx.x;                             // 1024 threads
  li[h] = leaf_idxs[h];
  __syncthreads();
  float m = -1e30f;
  for (int l = 0; l < NLEAF; ++l)
    m = fmaxf(m, node_h[(size_t)li[l] * HID + h]);
  fs[h] = m;
  __syncthreads();
  for (int c = 0; c < 4; ++c) {
    red[h] = Wout[c * HID + h] * fs[h];
    __syncthreads();
    for (int s = HID / 2; s > 0; s >>= 1) {
      if (h < s) red[h] += red[h + s];
      __syncthreads();
    }
    if (h == 0) logits[c] = red[0] + bout[c];
    __syncthreads();
  }
  if (h == 0) {
    float mx = fmaxf(fmaxf(logits[0], logits[1]), fmaxf(logits[2], logits[3]));
    float e[4], s = 0.f;
    for (int c = 0; c < 4; ++c) { e[c] = __expf(logits[c] - mx); s += e[c]; }
    float loss = 0.f;
    for (int c = 0; c < 4; ++c) {
      float p = e[c] / s;
      out[c] = p;
      float d = y[c] - p;
      loss += d * d;
    }
    out[4] = loss;
  }
}

// ---------------------------------------------------------------- launch ---
extern "C" void kernel_launch(void* const* d_in, const int* in_sizes, int n_in,
                              void* d_out, int out_size, void* d_ws, size_t ws_size,
                              hipStream_t stream) {
  const float* x_word   = (const float*)d_in[0];
  const int*   x_index  = (const int*)d_in[1];
  const int*   parent   = (const int*)d_in[2];
  const int*   leaf_idx = (const int*)d_in[3];
  const float* y        = (const float*)d_in[4];
  const float* E        = (const float*)d_in[5];
  const float* W_z = (const float*)d_in[6],  *U_z = (const float*)d_in[7],  *b_z = (const float*)d_in[8];
  const float* W_r = (const float*)d_in[9],  *U_r = (const float*)d_in[10], *b_r = (const float*)d_in[11];
  const float* W_h = (const float*)d_in[12], *U_h = (const float*)d_in[13], *b_h = (const float*)d_in[14];
  const float* W_out = (const float*)d_in[15], *b_out = (const float*)d_in[16];

  char* p = (char*)d_ws;
  size_t o = 0;
  auto carve = [&](size_t bytes) { char* r = p + o; o += (bytes + 63) & ~size_t(63); return r; };
  half_t*   xe16      = (half_t*)carve((size_t)NNODE * HID * 2);
  half_t*   W16       = (half_t*)carve((size_t)3 * HID * HID * 2);
  half_t*   U16       = (half_t*)carve((size_t)3 * HID * HID * 2);
  float*    wzrh      = (float*) carve((size_t)3 * NNODE * HID * 4);
  float*    node_h    = (float*) carve((size_t)(NNODE + 1) * HID * 4);
  half_t*   node_h16  = (half_t*)carve((size_t)(NNODE + 1) * HID * 2);
  half_t*   rph16     = (half_t*)carve((size_t)NNODE * HID * 2);
  float*    zbuf      = (float*) carve((size_t)NNODE * HID * 4);
  int*      depth     = (int*)   carve(NNODE * 4);
  int*      level_n   = (int*)   carve(NNODE * 4);
  int*      level_off = (int*)   carve(NNODE * 4);
  int*      cursor    = (int*)   carve(NNODE * 4);
  int*      nodes_srt = (int*)   carve(65536 * 4);
  int*      nlevels   = (int*)   carve(64);
  unsigned* bars      = (unsigned*)carve(16384 * 4);

  init_ws_kernel<<<64, 256, 0, stream>>>(node_h, node_h16, bars);
  cvt_weights_kernel<<<2048, 256, 0, stream>>>(W_z, W_r, W_h, U_z, U_r, U_h, W16, U16);
  embed_kernel<<<NNODE, 256, 0, stream>>>(x_word, x_index, E, xe16);
  // 32x32 tiles: (4096/32) m-tiles x (3072/32) n-tiles, 8 waves/block.
  gemm_xew_kernel<<<(3 * HID / 32) * ((NNODE / 32) / 8), 256, 0, stream>>>(
      xe16, W16, b_z, b_r, b_h, wzrh);
  build_levels_kernel<<<1, 32, 0, stream>>>(parent, depth, level_n, level_off,
                                            cursor, nodes_srt, nlevels);
  tree_scan_kernel<<<SCAN_BLOCKS, 256, 3 * 16 * HID * sizeof(half_t), stream>>>(
      parent, wzrh, U16, node_h, node_h16, rph16, zbuf,
      level_n, level_off, nodes_srt, nlevels, bars, SCAN_BLOCKS);
  finalize_kernel<<<1, HID, 0, stream>>>(node_h, leaf_idx, W_out, b_out, y,
                                         (float*)d_out);
}
